// AMIPRouter_43241730736379
// MI455X (gfx1250) — matrix operand order repaired and checked by reading, not compiled, not run on hardware
//
#include <hip/hip_runtime.h>
#include <hip/hip_bf16.h>

// Problem constants (fixed by setup_inputs): B=4, S=2048, D=4096, K=8, M=S/2.
#define BB   4
#define SS   2048
#define DD   4096
#define KK   8
#define MM   1024          // masked tokens per batch
#define UU   1024          // unmasked tokens per batch (S - M)
#define BM   (BB*MM)       // 4096 total tokens
#define TWO_D (2*DD)       // 8192  (GEMM1 reduction dim, also GEMM2 reduction dim K*D/4)
#define HID  (DD/4)        // 1024 per-expert hidden

typedef __attribute__((ext_vector_type(16))) __bf16 v16bf;
typedef __attribute__((ext_vector_type(8)))  float  v8f;
typedef __attribute__((ext_vector_type(4)))  int    v4i;
typedef unsigned short u16;
typedef unsigned int   u32;
typedef unsigned long long u64;

#define AS1 __attribute__((address_space(1)))
#define AS3 __attribute__((address_space(3)))

#if defined(__has_builtin)
# if __has_builtin(__builtin_amdgcn_global_load_async_to_lds_b128)
#  define USE_ASYNC_LDS 1
# endif
# if __has_builtin(__builtin_amdgcn_s_wait_asynccnt)
#  define HAVE_WAIT_ASYNC_BUILTIN 1
# endif
#endif

__device__ __forceinline__ void wait_async0() {
#if defined(HAVE_WAIT_ASYNC_BUILTIN)
    __builtin_amdgcn_s_wait_asynccnt(0);
#elif defined(USE_ASYNC_LDS)
    asm volatile("s_wait_asynccnt 0x0" ::: "memory");
#endif
}

#if defined(USE_ASYNC_LDS)
// 16-byte Global->LDS DMA (tracked by ASYNCcnt, no VGPR round-trip).
// Pointer casts go through integers: as1 keeps the 64-bit global address,
// as3 truncates the generic LDS pointer to its 32-bit LDS offset.
__device__ __forceinline__ void async_copy16(const u16* g, u16* l) {
    __builtin_amdgcn_global_load_async_to_lds_b128(
        (AS1 v4i*)(u64)g,
        (AS3 v4i*)(u32)(u64)l,
        0, 0);
}
#endif

union FragAB { u32 u[8]; v16bf v; };

__device__ __forceinline__ u16 f2bf(float f) {
    union { float f; u32 u; } v; v.f = f;
    u32 r = v.u + 0x7fffu + ((v.u >> 16) & 1u);   // round-to-nearest-even
    return (u16)(r >> 16);
}

// ---------------------------------------------------------------------------
// Kernel 0: zero the whole output tensor (non-masked rows must stay zero).
// ---------------------------------------------------------------------------
__global__ __launch_bounds__(256) void zero_kernel(float4* __restrict__ p, size_t n4) {
    size_t i = (size_t)blockIdx.x * blockDim.x + threadIdx.x;
    size_t stride = (size_t)gridDim.x * blockDim.x;
    float4 z = make_float4(0.f, 0.f, 0.f, 0.f);
    for (; i < n4; i += stride) p[i] = z;
}

// ---------------------------------------------------------------------------
// Kernel 1: per masked token -> neighbor average, routing softmax gate,
//           bf16 x = [h_avg, h_mask], cnt flag. One 256-thread block / token.
// ---------------------------------------------------------------------------
__device__ __forceinline__ bool sorted_contains(const int* __restrict__ arr, int n, int key) {
    int lo = 0, hi = n - 1;
    while (lo <= hi) {
        int mid = (lo + hi) >> 1;
        int v = arr[mid];
        if (v == key) return true;
        if (v < key) lo = mid + 1; else hi = mid - 1;
    }
    return false;
}

__global__ __launch_bounds__(256) void prep_kernel(
        const float* __restrict__ h_L, const float* __restrict__ W_r,
        const float* __restrict__ b_r, const int* __restrict__ mask_idx,
        const int* __restrict__ unmask_idx, const int* __restrict__ range_r_p,
        float* __restrict__ gate, int* __restrict__ cntf, u16* __restrict__ xbf)
{
    const int bm  = blockIdx.x;
    const int b   = bm >> 10;
    const int m   = bm & (MM - 1);
    const int tid = threadIdx.x;

    __shared__ int   spos[32];
    __shared__ int   scnt;
    __shared__ float slog[KK];
    __shared__ float sbuf[256];

    const int tgt = mask_idx[b * MM + m];
    int r = range_r_p[0];
    if (r > 16) r = 16;
    if (r < 0)  r = 0;

    if (tid == 0) scnt = 0;
    __syncthreads();

    if (tid < 2 * r) {
        int off = (tid < r) ? (tid - r) : (tid - r + 1);     // skip 0
        int pos = tgt + off;
        bool ok = (pos >= 0) && (pos < SS) &&
                  sorted_contains(unmask_idx + b * UU, UU, pos);
        spos[tid] = ok ? pos : -1;
        if (ok) atomicAdd(&scnt, 1);
    }
    __syncthreads();

    const int cnt = scnt;
    const float invc = 1.0f / (float)(cnt > 0 ? cnt : 1);
    const float* hb = h_L + (size_t)b * SS * DD;

    float gp[KK];
#pragma unroll
    for (int k = 0; k < KK; ++k) gp[k] = 0.f;

#pragma unroll 4
    for (int i = 0; i < DD / 256; ++i) {
        int d = tid + 256 * i;
        float s = 0.f;
        for (int nn = 0; nn < 2 * r; ++nn) {
            int p = spos[nn];
            if (p >= 0) s += hb[(size_t)p * DD + d];
        }
        float avg = s * invc;
        float hm  = hb[(size_t)tgt * DD + d];
        xbf[(size_t)bm * TWO_D + d]      = f2bf(avg);
        xbf[(size_t)bm * TWO_D + DD + d] = f2bf(hm);
#pragma unroll
        for (int k = 0; k < KK; ++k) gp[k] += hm * W_r[d * KK + k];
    }

    // deterministic tree reduction of the 8 routing logits
    for (int k = 0; k < KK; ++k) {
        sbuf[tid] = gp[k];
        __syncthreads();
        for (int st = 128; st > 0; st >>= 1) {
            if (tid < st) sbuf[tid] += sbuf[tid + st];
            __syncthreads();
        }
        if (tid == 0) slog[k] = sbuf[0] + b_r[k];
        __syncthreads();
    }

    if (tid == 0) {
        float mx = slog[0];
#pragma unroll
        for (int k = 1; k < KK; ++k) mx = fmaxf(mx, slog[k]);
        float e[KK], se = 0.f;
#pragma unroll
        for (int k = 0; k < KK; ++k) { e[k] = __expf(slog[k] - mx); se += e[k]; }
        float inv = 1.0f / se;
#pragma unroll
        for (int k = 0; k < KK; ++k) gate[bm * KK + k] = e[k] * inv;
        cntf[bm] = (cnt > 0) ? 1 : 0;
    }
}

// ---------------------------------------------------------------------------
// WMMA fragment gathers (16-bit operand layouts per CDNA5 ISA 7.12.2)
// A tile in LDS: row-major [128][LDA] bf16.   B tile in LDS: transposed
// [n][LDB] bf16 so (k,k+1) pairs are contiguous 32-bit words.
// LDA*2 = 80 bytes (multiple of 16) keeps b128 alignment for DMA/ds ops.
// ---------------------------------------------------------------------------
#define LDA 40
#define LDB 40

__device__ __forceinline__ void load_fragA(FragAB& f, const u16* sA, int mbase, int lr, int hi) {
#pragma unroll
    for (int v = 0; v < 8; ++v) {
        int k = ((v < 4) ? (2 * v) : (2 * v + 8)) + 8 * hi;   // K pair start
        f.u[v] = *(const u32*)&sA[(mbase + lr) * LDA + k];
    }
}
__device__ __forceinline__ void load_fragB(FragAB& f, const u16* sB, int nbase, int lr, int hi) {
#pragma unroll
    for (int v = 0; v < 8; ++v) {
        int k = ((v < 4) ? (2 * v) : (2 * v + 8)) + 8 * hi;   // K pair start
        f.u[v] = *(const u32*)&sB[(nbase + lr) * LDB + k];
    }
}

// Stage a 128x32 bf16 A tile (row-major source, row stride TWO_D) into sA.
// Uses the gfx1250 async Global->LDS DMA path when available; otherwise a
// batched b128 copy with both loads issued before the stores.
__device__ __forceinline__ void stage_A(u16* __restrict__ sA,
                                        const u16* __restrict__ src,  // + bm*TWO_D + kk
                                        int tid)
{
    const int c0 = tid, c1 = tid + 256;        // two 16B chunks of 512
    const int r0 = c0 >> 2, f0 = (c0 & 3) * 8; // row, elem offset
    const int r1 = c1 >> 2, f1 = (c1 & 3) * 8;
    const u16* g0 = src + (size_t)r0 * TWO_D + f0;
    const u16* g1 = src + (size_t)r1 * TWO_D + f1;
#if defined(USE_ASYNC_LDS)
    async_copy16(g0, &sA[r0 * LDA + f0]);
    async_copy16(g1, &sA[r1 * LDA + f1]);
#else
    uint4 t0 = *(const uint4*)g0;              // both loads issued before stores
    uint4 t1 = *(const uint4*)g1;
    *(uint4*)&sA[r0 * LDA + f0] = t0;
    *(uint4*)&sA[r1 * LDA + f1] = t1;
#endif
}

// Stage a 32x128 fp32 weight tile transposed+converted into sB[n][k] bf16.
// All 4 global b128 loads are issued before any LDS store.
__device__ __forceinline__ void stage_B(u16* __restrict__ sB,
                                        const float* __restrict__ wrow, // 16 consecutive floats
                                        int fstage, int nstage)
{
    float4 wv[4];
#pragma unroll
    for (int i = 0; i < 4; ++i) wv[i] = *(const float4*)&wrow[i * 4];
    const float* wf = (const float*)wv;
#pragma unroll
    for (int i = 0; i < 16; ++i)
        sB[(nstage + i) * LDB + fstage] = f2bf(wf[i]);
}

// ---------------------------------------------------------------------------
// Kernel 2: GEMM1  C[BM,8192] = Xbf16[BM,8192] @ W1cat(bf16) ; epilogue:
//           + b1, exact GELU, * gate[row, expert(col)], store bf16 hidg.
// ---------------------------------------------------------------------------
__global__ __launch_bounds__(256) void gemm1_kernel(
        const u16* __restrict__ xbf, const float* __restrict__ W1,
        const float* __restrict__ b1, const float* __restrict__ gate,
        u16* __restrict__ hidg)
{
    __shared__ u16 sA[128 * LDA];
    __shared__ u16 sB[128 * LDB];

    const int bn  = blockIdx.x * 128;           // output col base (0..8191)
    const int bm  = blockIdx.y * 128;           // output row base (0..4095)
    const int tid = threadIdx.x;
    const int lane = tid & 31, wid = tid >> 5;
    const int wm = (wid >> 1) * 32;             // wave row offset within block
    const int wn = (wid & 1) * 64;              // wave col offset within block
    const int lr = lane & 15, hi = lane >> 4;

    const int kE = bn >> 10;                    // expert of this col tile
    const int h0 = bn & (HID - 1);

    v8f acc[2][4];
#pragma unroll
    for (int i = 0; i < 2; ++i)
#pragma unroll
        for (int j = 0; j < 4; ++j) acc[i][j] = (v8f){0.f,0.f,0.f,0.f,0.f,0.f,0.f,0.f};

    const int fstage = tid >> 3;                // 0..31 : W1 row within k-slab
    const int nstage = (tid & 7) * 16;          // col start within tile

    for (int kk = 0; kk < TWO_D; kk += 32) {
        __syncthreads();
        stage_A(sA, xbf + (size_t)bm * TWO_D + kk, tid);
        stage_B(sB, &W1[((size_t)kE * TWO_D + kk + fstage) * HID + h0 + nstage],
                fstage, nstage);
        if (kk + 32 < TWO_D)
            __builtin_prefetch(&W1[((size_t)kE * TWO_D + kk + 32 + fstage) * HID + h0 + nstage], 0, 3);
        wait_async0();
        __syncthreads();

        FragAB af[2], bf_[4];
#pragma unroll
        for (int i = 0; i < 2; ++i) load_fragA(af[i], sA, wm + i * 16, lr, hi);
#pragma unroll
        for (int j = 0; j < 4; ++j) load_fragB(bf_[j], sB, wn + j * 16, lr, hi);
#pragma unroll
        for (int i = 0; i < 2; ++i)
#pragma unroll
            for (int j = 0; j < 4; ++j)
                acc[i][j] = __builtin_amdgcn_wmma_f32_16x16x32_bf16(
                    false, af[i].v, false, bf_[j].v, (short)0, acc[i][j], false, false);
    }

    // epilogue: bias + exact GELU + gate, store bf16
#pragma unroll
    for (int i = 0; i < 2; ++i) {
#pragma unroll
        for (int j = 0; j < 4; ++j) {
#pragma unroll
            for (int e = 0; e < 8; ++e) {
                int mloc = wm + i * 16 + (hi ? 8 : 0) + e;
                int nloc = wn + j * 16 + lr;
                int row = bm + mloc;
                int col = bn + nloc;
                float v = acc[i][j][e] + b1[col];
                float g = 0.5f * v * (1.0f + erff(v * 0.70710678118654752f));
                g *= gate[row * KK + kE];
                hidg[(size_t)row * TWO_D + col] = f2bf(g);
            }
        }
    }
}

// ---------------------------------------------------------------------------
// Kernel 3: GEMM2  E[BM,4096] = hidg @ W2flat(bf16) ; epilogue:
//           + sum_k gate*b2, scatter rows to d_out at mask positions.
// ---------------------------------------------------------------------------
__global__ __launch_bounds__(256) void gemm2_kernel(
        const u16* __restrict__ hidg, const float* __restrict__ W2,
        const float* __restrict__ b2, const float* __restrict__ gate,
        const int* __restrict__ mask_idx, float* __restrict__ out)
{
    __shared__ u16 sA[128 * LDA];
    __shared__ u16 sB[128 * LDB];

    const int bn  = blockIdx.x * 128;           // output col base (0..4095)
    const int bm  = blockIdx.y * 128;           // output row base (0..4095)
    const int tid = threadIdx.x;
    const int lane = tid & 31, wid = tid >> 5;
    const int wm = (wid >> 1) * 32;
    const int wn = (wid & 1) * 64;
    const int lr = lane & 15, hi = lane >> 4;

    v8f acc[2][4];
#pragma unroll
    for (int i = 0; i < 2; ++i)
#pragma unroll
        for (int j = 0; j < 4; ++j) acc[i][j] = (v8f){0.f,0.f,0.f,0.f,0.f,0.f,0.f,0.f};

    const int fstage = tid >> 3;
    const int nstage = (tid & 7) * 16;

    for (int kk = 0; kk < TWO_D; kk += 32) {
        __syncthreads();
        stage_A(sA, hidg + (size_t)bm * TWO_D + kk, tid);
        stage_B(sB, &W2[(size_t)(kk + fstage) * DD + bn + nstage], fstage, nstage);
        if (kk + 32 < TWO_D)
            __builtin_prefetch(&W2[(size_t)(kk + 32 + fstage) * DD + bn + nstage], 0, 3);
        wait_async0();
        __syncthreads();

        FragAB af[2], bf_[4];
#pragma unroll
        for (int i = 0; i < 2; ++i) load_fragA(af[i], sA, wm + i * 16, lr, hi);
#pragma unroll
        for (int j = 0; j < 4; ++j) load_fragB(bf_[j], sB, wn + j * 16, lr, hi);
#pragma unroll
        for (int i = 0; i < 2; ++i)
#pragma unroll
            for (int j = 0; j < 4; ++j)
                acc[i][j] = __builtin_amdgcn_wmma_f32_16x16x32_bf16(
                    false, af[i].v, false, bf_[j].v, (short)0, acc[i][j], false, false);
    }

    // epilogue: gate-weighted bias, scatter to out[b, mask[b,m], col]
#pragma unroll
    for (int i = 0; i < 2; ++i) {
#pragma unroll
        for (int e = 0; e < 8; ++e) {
            int mloc = wm + i * 16 + (hi ? 8 : 0) + e;
            int row  = bm + mloc;
            int bIdx = row >> 10;
            int mIdx = row & (MM - 1);
            int tgt  = mask_idx[bIdx * MM + mIdx];
            float gr[KK];
#pragma unroll
            for (int k = 0; k < KK; ++k) gr[k] = gate[row * KK + k];
            float* orow = out + ((size_t)bIdx * SS + tgt) * DD;
#pragma unroll
            for (int j = 0; j < 4; ++j) {
                int col = bn + wn + j * 16 + lr;
                float bias = 0.f;
#pragma unroll
                for (int k = 0; k < KK; ++k) bias += gr[k] * b2[k * DD + col];
                orow[col] = acc[i][j][e] + bias;
            }
        }
    }
}

// ---------------------------------------------------------------------------
// Kernel 4: LayerNorm (eps 1e-5, no affine) in place on scattered rows;
//           rows with cnt==0 forced to zero.
// ---------------------------------------------------------------------------
__global__ __launch_bounds__(256) void ln_kernel(
        float* __restrict__ out, const int* __restrict__ mask_idx,
        const int* __restrict__ cntf)
{
    const int bm  = blockIdx.x;
    const int b   = bm >> 10;
    const int m   = bm & (MM - 1);
    const int tid = threadIdx.x;
    const int tgt = mask_idx[b * MM + m];
    float* row = out + ((size_t)b * SS + tgt) * DD;

    __shared__ float rs[256];
    __shared__ float rq[256];

    float v[DD / 256];
    float s = 0.f, q = 0.f;
#pragma unroll 4
    for (int i = 0; i < DD / 256; ++i) {
        float x = row[tid + 256 * i];
        v[i] = x; s += x; q += x * x;
    }
    rs[tid] = s; rq[tid] = q;
    __syncthreads();
    for (int st = 128; st > 0; st >>= 1) {
        if (tid < st) { rs[tid] += rs[tid + st]; rq[tid] += rq[tid + st]; }
        __syncthreads();
    }
    const float mean = rs[0] * (1.0f / DD);
    const float var  = rq[0] * (1.0f / DD) - mean * mean;
    const float rinv = rsqrtf(var + 1e-5f);
    const float keep = (cntf[bm] > 0) ? 1.0f : 0.0f;
#pragma unroll 4
    for (int i = 0; i < DD / 256; ++i)
        row[tid + 256 * i] = keep * (v[i] - mean) * rinv;
}

// ---------------------------------------------------------------------------
extern "C" void kernel_launch(void* const* d_in, const int* in_sizes, int n_in,
                              void* d_out, int out_size, void* d_ws, size_t ws_size,
                              hipStream_t stream) {
    const float* h_L        = (const float*)d_in[0];
    const float* W_r        = (const float*)d_in[1];
    const float* b_r        = (const float*)d_in[2];
    const float* W1         = (const float*)d_in[3];
    const float* b1         = (const float*)d_in[4];
    const float* W2         = (const float*)d_in[5];
    const float* b2         = (const float*)d_in[6];
    const int*   mask_idx   = (const int*)d_in[7];
    const int*   unmask_idx = (const int*)d_in[8];
    const int*   range_r    = (const int*)d_in[9];
    float* out = (float*)d_out;

    // workspace carve-out (256B aligned): gate, cnt, x_bf16, hidg_bf16
    char* w = (char*)d_ws;
    size_t off = 0;
    auto up = [](size_t x) { return (x + 255) & ~(size_t)255; };
    float* gate = (float*)(w + off); off = up(off + (size_t)BM * KK * sizeof(float));
    int*   cnt  = (int*)(w + off);   off = up(off + (size_t)BM * sizeof(int));
    u16*   xbf  = (u16*)(w + off);   off = up(off + (size_t)BM * TWO_D * sizeof(u16));
    u16*   hidg = (u16*)(w + off);   off = up(off + (size_t)BM * TWO_D * sizeof(u16));
    (void)ws_size; (void)in_sizes; (void)n_in; (void)out_size;

    const size_t n4 = (size_t)BB * SS * DD / 4;
    zero_kernel<<<4096, 256, 0, stream>>>((float4*)out, n4);
    prep_kernel<<<BM, 256, 0, stream>>>(h_L, W_r, b_r, mask_idx, unmask_idx,
                                        range_r, gate, cnt, xbf);
    gemm1_kernel<<<dim3(TWO_D / 128, BM / 128), 256, 0, stream>>>(xbf, W1, b1, gate, hidg);
    gemm2_kernel<<<dim3(DD / 128, BM / 128), 256, 0, stream>>>(hidg, W2, b2, gate, mask_idx, out);
    ln_kernel<<<BM, 256, 0, stream>>>(out, mask_idx, cnt);
}
